// PixelTransformerResnet_82712480186863
// MI455X (gfx1250) — matrix operand order repaired and checked by reading, not compiled, not run on hardware
//
#include <hip/hip_runtime.h>
#include <hip/hip_bf16.h>

// ---------------------------------------------------------------------------
// PixelTransformer block, MI455X (gfx1250, wave32, WMMA).
//  Stage 1 (x3): q/k/v = relu(bn(W_i @ x + b_i))        -> bf16 workspace
//  Stage 2     : 3x3-neighborhood attention + bn + relu -> bf16 workspace
//  Stage 3     : out = relu(bn(W4 @ y + b4) + x)        -> f32 d_out
// GEMMs use v_wmma_f32_16x16x32_bf16 with f32 accumulation (K=256).
// 512 threads / 16 waves per block; wave tile 16(M) x 128(N) => 64 acc VGPRs,
// no accumulator spills.
// Workspace requirement: 4 * 8*256*4096 * 2 bytes = 64 MB.
// ---------------------------------------------------------------------------

typedef __bf16 bf16;
typedef __attribute__((ext_vector_type(16))) __bf16 v16bf;
typedef __attribute__((ext_vector_type(8)))  __bf16 v8bf;
typedef __attribute__((ext_vector_type(8)))  float  v8f;

#define CCH    256          // channels (in == mid == out)
#define HWPX   4096         // 64*64 pixels per image
#define IMG_W  64
#define NIMG   8
#define TILE_P 128          // pixel tile per block
#define KCHUNK 64           // K staged per LDS pass
#define KP     72           // padded LDS pitch in halves (row = 144 B, 16B-aligned)
#define NTHR   512          // 16 waves

static const size_t TOTPX = (size_t)NIMG * CCH * HWPX;   // elems per tensor

union ABu { v16bf v; v8bf h[2]; };

__device__ __forceinline__ v16bf ld_frag(const bf16* p0, const bf16* p1) {
  ABu u;
  u.h[0] = *(const v8bf*)p0;
  u.h[1] = *(const v8bf*)p1;
  return u.v;
}

// ---------------------------------------------------------------------------
// GEMM + bias + BN (+ReLU | +residual+ReLU) kernel.
//   out[o,p] = act( (W@X + bias - mean)*g*rsqrt(var+eps) + beta [ + resid ] )
// Block: 512 threads (16 waves). Tile: 256 (M) x 128 (P). Wave: 16(M) x 128(P).
// ---------------------------------------------------------------------------
template<bool IN_BF16, bool RESIDUAL>
__global__ void __launch_bounds__(NTHR, 1)
gemm_bn_kernel(const float* __restrict__ Wt,
               const void*  __restrict__ Xin,
               const float* __restrict__ bias,
               const float* __restrict__ bng,
               const float* __restrict__ bnb,
               const float* __restrict__ bnm,
               const float* __restrict__ bnv,
               const float* __restrict__ resid,
               void*        __restrict__ Out)
{
  __shared__ bf16  sW[CCH][KP];       // [o][k]  36.9 KB
  __shared__ bf16  sX[TILE_P][KP];    // [p][k]  18.4 KB
  __shared__ float sAlpha[CCH], sBeta[CCH];

  const int tid  = threadIdx.x;
  const int lane = tid & 31;
  const int wave = tid >> 5;
  const int b    = blockIdx.y;
  const int pix0 = blockIdx.x * TILE_P;
  const int m0   = wave * 16;          // 16 output channels per wave

  // fold BN + conv-bias into alpha/beta per output channel
  if (tid < CCH) {
    float inv = rsqrtf(bnv[tid] + 1e-5f);
    float al  = bng[tid] * inv;
    sAlpha[tid] = al;
    sBeta[tid]  = bnb[tid] + (bias[tid] - bnm[tid]) * al;
  }

  const size_t imgoff = (size_t)b * CCH * HWPX;
  const float* Xf = (const float*)Xin + imgoff;
  const bf16*  Xb = (const bf16*)Xin  + imgoff;

  v8f acc[8];
  const v8f vzero = {};
  #pragma unroll
  for (int nn = 0; nn < 8; ++nn) acc[nn] = vzero;

  const int kbA  = (lane < 16) ? 0 : 8;   // A frag: lane<16 -> K 0..7,16..23
  const int kbB  = (lane < 16) ? 0 : 8;   // B frag: lane<16 -> K 0..7,16..23
  const int mrow = lane & 15;

  for (int kc = 0; kc < CCH; kc += KCHUNK) {
    __syncthreads();
    // stage weights: 256 x 64 (f32 -> bf16)
    #pragma unroll 4
    for (int i = 0; i < (CCH * KCHUNK) / NTHR; ++i) {   // 32 iters
      int idx = i * NTHR + tid;
      int m = idx >> 6;        // /64
      int k = idx & 63;
      sW[m][k] = (bf16)Wt[m * CCH + kc + k];
    }
    // stage activations: 64(K) x 128(P), transposed to [p][k]
    #pragma unroll 4
    for (int i = 0; i < (KCHUNK * TILE_P) / NTHR; ++i) { // 16 iters
      int idx = i * NTHR + tid;
      int k = idx >> 7;        // /128
      int p = idx & 127;
      float xv;
      if (IN_BF16) xv = (float)Xb[(size_t)(kc + k) * HWPX + pix0 + p];
      else         xv =        Xf[(size_t)(kc + k) * HWPX + pix0 + p];
      sX[p][k] = (bf16)xv;
    }
    __syncthreads();

    #pragma unroll
    for (int ks = 0; ks < 2; ++ks) {          // two K=32 steps per chunk
      const int kw = ks * 32;
      const bf16* pa = &sW[m0 + mrow][kw + kbA];
      const v16bf afr = ld_frag(pa, pa + 16);
      #pragma unroll
      for (int nn = 0; nn < 8; ++nn) {
        const bf16* pb = &sX[nn * 16 + mrow][kw + kbB];
        const v16bf bfr = ld_frag(pb, pb + 16);
        acc[nn] = __builtin_amdgcn_wmma_f32_16x16x32_bf16(
            false, afr, false, bfr, (short)0, acc[nn], false, false);
      }
    }
  }

  // Epilogue. f32 C/D layout: VGPR r -> M = r (lanes 0-15) / r+8 (lanes 16-31),
  // N = lane & 15.
  const int nlo = lane & 15;
  const int mhi = (lane < 16) ? 0 : 8;
  #pragma unroll
  for (int nn = 0; nn < 8; ++nn) {
    const int p = pix0 + nn * 16 + nlo;
    #pragma unroll
    for (int r = 0; r < 8; ++r) {
      const int o = m0 + mhi + r;
      float y = acc[nn][r] * sAlpha[o] + sBeta[o];
      const size_t off = imgoff + (size_t)o * HWPX + p;
      if (RESIDUAL) {
        y = fmaxf(y + resid[off], 0.0f);
        ((float*)Out)[off] = y;
      } else {
        y = fmaxf(y, 0.0f);
        ((bf16*)Out)[off] = (bf16)y;
      }
    }
  }
}

// ---------------------------------------------------------------------------
// Attention: one thread per pixel; C-loop loads are contiguous across lanes.
// q/k/v (50 MB bf16) live in the 192 MB L2, so 9x neighbor re-reads are cheap.
// OOB neighbors: logit contribution 0 (kept in softmax denom, as reference)
// and value contribution 0.
// ---------------------------------------------------------------------------
__global__ void attention_bn_kernel(const bf16* __restrict__ q,
                                    const bf16* __restrict__ k,
                                    const bf16* __restrict__ v,
                                    const float* __restrict__ bng,
                                    const float* __restrict__ bnb,
                                    const float* __restrict__ bnm,
                                    const float* __restrict__ bnv,
                                    bf16* __restrict__ yout)
{
  __shared__ float sAlpha[CCH], sBeta[CCH];
  const int tid = threadIdx.x;
  if (tid < CCH) {
    float inv = rsqrtf(bnv[tid] + 1e-5f);
    float al  = bng[tid] * inv;
    sAlpha[tid] = al;
    sBeta[tid]  = bnb[tid] - bnm[tid] * al;
  }
  __syncthreads();

  const int b  = blockIdx.y;
  const int p  = blockIdx.x * blockDim.x + tid;
  const int py = p >> 6;
  const int px = p & (IMG_W - 1);
  const size_t base = (size_t)b * CCH * HWPX + p;
  const bf16* qb = q + base;
  const bf16* kb = k + base;
  const bf16* vb = v + base;

  long  offc[9];
  float msk[9];
  #pragma unroll
  for (int n = 0; n < 9; ++n) {
    int dy = n / 3 - 1, dx = n % 3 - 1;
    bool ok = (py + dy >= 0) && (py + dy < IMG_W) &&
              (px + dx >= 0) && (px + dx < IMG_W);
    offc[n] = ok ? (long)(dy * IMG_W + dx) : 0;   // clamped -> always in-bounds
    msk[n]  = ok ? 1.0f : 0.0f;
  }

  float lg[9];
  #pragma unroll
  for (int n = 0; n < 9; ++n) lg[n] = 0.0f;

  for (int c = 0; c < CCH; ++c) {
    const long co = (long)c * HWPX;
    const float qv = (float)qb[co];
    #pragma unroll
    for (int n = 0; n < 9; ++n)
      lg[n] += qv * msk[n] * (float)kb[co + offc[n]];
  }

  // softmax over 9 (scale = 1/sqrt(256) = 1/16)
  #pragma unroll
  for (int n = 0; n < 9; ++n) lg[n] *= 0.0625f;
  float mx = lg[0];
  #pragma unroll
  for (int n = 1; n < 9; ++n) mx = fmaxf(mx, lg[n]);
  float at[9], s = 0.0f;
  #pragma unroll
  for (int n = 0; n < 9; ++n) { at[n] = __expf(lg[n] - mx); s += at[n]; }
  const float rs = 1.0f / s;
  #pragma unroll
  for (int n = 0; n < 9; ++n) at[n] *= rs * msk[n];

  for (int c = 0; c < CCH; ++c) {
    const long co = (long)c * HWPX;
    float a = 0.0f;
    #pragma unroll
    for (int n = 0; n < 9; ++n)
      a += at[n] * (float)vb[co + offc[n]];
    float y = a * sAlpha[c] + sBeta[c];
    y = fmaxf(y, 0.0f);
    yout[base + (size_t)c * HWPX] = (bf16)y;
  }
}

// ---------------------------------------------------------------------------
extern "C" void kernel_launch(void* const* d_in, const int* in_sizes, int n_in,
                              void* d_out, int out_size, void* d_ws, size_t ws_size,
                              hipStream_t stream)
{
  (void)in_sizes; (void)n_in; (void)out_size; (void)ws_size;

  const float* x  = (const float*)d_in[0];
  const float* W1 = (const float*)d_in[1];
  const float* b1 = (const float*)d_in[2];
  const float* W2 = (const float*)d_in[3];
  const float* b2 = (const float*)d_in[4];
  const float* W3 = (const float*)d_in[5];
  const float* b3 = (const float*)d_in[6];
  const float* W4 = (const float*)d_in[7];
  const float* b4 = (const float*)d_in[8];
  const float* bg = (const float*)d_in[9];
  const float* bb = (const float*)d_in[10];
  const float* bm = (const float*)d_in[11];
  const float* bv = (const float*)d_in[12];

  bf16* qws = (bf16*)d_ws;
  bf16* kws = qws + TOTPX;
  bf16* vws = kws + TOTPX;
  bf16* yws = vws + TOTPX;   // total ws use: 64 MB

  const dim3 gblk(NTHR);
  const dim3 grd(HWPX / TILE_P, NIMG);

  gemm_bn_kernel<false, false><<<grd, gblk, 0, stream>>>(
      W1, x, b1, bg + 0 * CCH, bb + 0 * CCH, bm + 0 * CCH, bv + 0 * CCH,
      nullptr, qws);
  gemm_bn_kernel<false, false><<<grd, gblk, 0, stream>>>(
      W2, x, b2, bg + 1 * CCH, bb + 1 * CCH, bm + 1 * CCH, bv + 1 * CCH,
      nullptr, kws);
  gemm_bn_kernel<false, false><<<grd, gblk, 0, stream>>>(
      W3, x, b3, bg + 2 * CCH, bb + 2 * CCH, bm + 2 * CCH, bv + 2 * CCH,
      nullptr, vws);

  attention_bn_kernel<<<dim3(HWPX / 256, NIMG), dim3(256), 0, stream>>>(
      qws, kws, vws, bg + 3 * CCH, bb + 3 * CCH, bm + 3 * CCH, bv + 3 * CCH,
      yws);

  gemm_bn_kernel<true, true><<<grd, gblk, 0, stream>>>(
      W4, yws, b4, bg + 4 * CCH, bb + 4 * CCH, bm + 4 * CCH, bv + 4 * CCH,
      x, d_out);
}
